// MapGraphModel_84224308675220
// MI455X (gfx1250) — compile-verified, matrix operand order checked by reading, not run on hardware
//
#include <hip/hip_runtime.h>

// ---------------- model dims ----------------
#define BB   32
#define NN   512
#define HH   64
#define WWD  64
#define DD   128
#define GEc  64
#define NTOT (BB*NN)        // 16384
#define NE   (BB*8192)      // 262144

typedef __bf16 bf16_t;
typedef __attribute__((ext_vector_type(16))) __bf16 v16bf;
typedef __attribute__((ext_vector_type(8)))  __bf16 v8bf;
typedef __attribute__((ext_vector_type(8)))  float  v8f;

// WMMA 16-bit A/B K mapping: VGPR v (0..7), element j (0..1), khalf = lane>>4
__device__ __forceinline__ int wmma_k0(int v, int khalf) {
    return ((v < 4) ? 0 : 16) + khalf * 8 + 2 * (v & 3);
}

// Load one 16-bit A fragment: two contiguous 8x bf16 runs (16B each) per lane.
__device__ __forceinline__ v16bf load_a16(const bf16_t* p, int khalf, bool ok) {
    v16bf a;
    if (ok) {
        v8bf lo = *(const v8bf*)(p + khalf * 8);
        v8bf hi = *(const v8bf*)(p + 16 + khalf * 8);
#pragma unroll
        for (int i = 0; i < 8; ++i) { a[i] = lo[i]; a[8 + i] = hi[i]; }
    } else {
#pragma unroll
        for (int i = 0; i < 16; ++i) a[i] = (bf16_t)0.f;
    }
    return a;
}

// ---------------- small utility kernels ----------------
__global__ void fillf_kernel(float* p, float v, long long n) {
    long long i = (long long)blockIdx.x * blockDim.x + threadIdx.x;
    if (i < n) p[i] = v;
}
__global__ void filli_kernel(int* p, int v, long long n) {
    long long i = (long long)blockIdx.x * blockDim.x + threadIdx.x;
    if (i < n) p[i] = v;
}
__global__ void f32_to_bf16_kernel(const float* __restrict__ in, bf16_t* __restrict__ out, long long n) {
    long long i = (long long)blockIdx.x * blockDim.x + threadIdx.x;
    if (i < n) out[i] = (bf16_t)in[i];
}

// ---------------- weight packing (f32 -> per-lane bf16 WMMA-B layout) ----------------
__global__ void pack_b_bf16(const float* __restrict__ W, bf16_t* __restrict__ out, int K, int N) {
    int lane = threadIdx.x;
    int NT = N >> 4;
    int nt = blockIdx.x % NT;
    int kt = blockIdx.x / NT;
    int n = nt * 16 + (lane & 15);
    int khalf = lane >> 4;
    bf16_t* o = out + ((long long)blockIdx.x * 32 + lane) * 16;
#pragma unroll
    for (int v = 0; v < 8; ++v) {
        int k0 = kt * 32 + wmma_k0(v, khalf);
        o[2 * v + 0] = (bf16_t)W[(long long)(k0 + 0) * N + n];
        o[2 * v + 1] = (bf16_t)W[(long long)(k0 + 1) * N + n];
    }
}

// Conv weight OIHW [Cout,Cin,3,3] -> tiles indexed ((ky*3+kx)*CT + ct)*NT + nt
__global__ void pack_conv_bf16(const float* __restrict__ w, bf16_t* __restrict__ out, int Cin, int Cout) {
    int lane = threadIdx.x;
    int NT = Cout >> 4, CT = Cin >> 5;
    int nt = blockIdx.x % NT;
    int t9 = blockIdx.x / NT;
    int ct = t9 % CT;
    int kk = t9 / CT;
    int ky = kk / 3, kx = kk % 3;
    int n = nt * 16 + (lane & 15);
    int khalf = lane >> 4;
    bf16_t* o = out + ((long long)blockIdx.x * 32 + lane) * 16;
#pragma unroll
    for (int v = 0; v < 8; ++v) {
        int k0 = wmma_k0(v, khalf);
        int c0 = ct * 32 + k0;
        o[2 * v + 0] = (bf16_t)w[(((long long)n * Cin + c0 + 0) * 3 + ky) * 3 + kx];
        o[2 * v + 1] = (bf16_t)w[(((long long)n * Cin + c0 + 1) * 3 + ky) * 3 + kx];
    }
}

// ---------------- WMMA GEMM: Y[M,N] = X[M,K] @ W (packed); 4 N-tiles per wave ----------------
__global__ __launch_bounds__(256)
void gemm_wmma(const bf16_t* __restrict__ X, const bf16_t* __restrict__ Wp,
               float* __restrict__ Y, int M, int K, int N) {
    int lane = threadIdx.x & 31;
    int NT = N >> 4;
    int NG = NT >> 2;                       // groups of 4 n-tiles
    long long wave = (long long)blockIdx.x * 8 + (threadIdx.x >> 5);
    int ng = (int)(wave % NG);
    int mt = (int)(wave / NG);
    int mlane = lane & 15, khalf = lane >> 4;
    const bf16_t* xrow = X + (long long)(mt * 16 + mlane) * K;
    v8f acc[4];
#pragma unroll
    for (int t = 0; t < 4; ++t)
#pragma unroll
        for (int i = 0; i < 8; ++i) acc[t][i] = 0.f;
    int KT = K >> 5;
    for (int kt = 0; kt < KT; ++kt) {
        v16bf a = load_a16(xrow + kt * 32, khalf, true);
#pragma unroll
        for (int t = 0; t < 4; ++t) {
            v16bf b = *(const v16bf*)(Wp + ((long long)(kt * NT + ng * 4 + t) * 32 + lane) * 16);
            acc[t] = __builtin_amdgcn_wmma_f32_16x16x32_bf16(false, a, false, b, (short)0, acc[t], false, false);
        }
    }
    int mb = mt * 16 + (khalf << 3);
#pragma unroll
    for (int t = 0; t < 4; ++t) {
        int n = ng * 64 + t * 16 + mlane;
#pragma unroll
        for (int r = 0; r < 8; ++r) Y[(long long)(mb + r) * N + n] = acc[t][r];
    }
}

// ---------------- graph kernels ----------------
__global__ void edge_deg(const int* __restrict__ src, const int* __restrict__ dst,
                         float* degOut, float* degIn, int ne) {
    int i = blockIdx.x * blockDim.x + threadIdx.x;
    if (i >= ne) return;
    atomicAdd(&degOut[src[i]], 1.f);
    atomicAdd(&degIn[dst[i]], 1.f);
}

__global__ void scatter_edges(const float* __restrict__ h, const int* __restrict__ src,
                              const int* __restrict__ dst, const float* __restrict__ degOut,
                              float* __restrict__ agg, int F, long long total) {
    long long i = (long long)blockIdx.x * blockDim.x + threadIdx.x;
    if (i >= total) return;
    int f = (int)(i % F);
    long long e = i / F;
    int s = src[e], d = dst[e];
    float sc = rsqrtf(fmaxf(degOut[s], 1.f));
    atomicAdd(&agg[(long long)d * F + f], h[(long long)s * F + f] * sc);
}

// relu, bf16 output (feeds next WMMA GEMM)
__global__ void scale_relu_bf16(const float* __restrict__ agg, const float* __restrict__ degIn,
                                bf16_t* __restrict__ out, int F, long long total) {
    long long i = (long long)blockIdx.x * blockDim.x + threadIdx.x;
    if (i >= total) return;
    long long row = i / F;
    float v = agg[i] * rsqrtf(fmaxf(degIn[row], 1.f));
    out[i] = (bf16_t)fmaxf(v, 0.f);
}

// tanh, f32 output (g: used for agent features and rep-map gather)
__global__ void scale_tanh_f32(const float* __restrict__ agg, const float* __restrict__ degIn,
                               float* __restrict__ out, int F, long long total) {
    long long i = (long long)blockIdx.x * blockDim.x + threadIdx.x;
    if (i >= total) return;
    long long row = i / F;
    float v = agg[i] * rsqrtf(fmaxf(degIn[row], 1.f));
    out[i] = tanhf(v);
}

// rep_map NHWC [B,H,W,64] in bf16
__global__ void build_repmap(const float* __restrict__ g, const int* __restrict__ mp,
                             bf16_t* __restrict__ out, long long total) {
    long long i = (long long)blockIdx.x * blockDim.x + threadIdx.x;
    if (i >= total) return;
    int c = (int)(i & 63);
    long long p = i >> 6;        // b*4096 + y*64 + x
    int b = (int)(p >> 12);
    int node = mp[p];
    out[i] = (node >= 0) ? (bf16_t)g[((long long)b * NN + node) * GEc + c] : (bf16_t)0.f;
}

// ---------------- WMMA implicit-GEMM 3x3 conv (pad 1, stride 1, 64x64 maps) ----------------
// in: NHWC bf16 [B,64,64,Cin]; 4 N-tiles (16x64 out strip) per wave.
// outB: NHWC bf16 (act=relu) ; outF: NCHW f32 (act=tanh, final rm)
__global__ __launch_bounds__(256)
void conv3x3_wmma(const bf16_t* __restrict__ in, const bf16_t* __restrict__ wp,
                  const float* __restrict__ bias, bf16_t* __restrict__ outB,
                  float* __restrict__ outF, int Cin, int Cout, int act, int nchwF) {
    int lane = threadIdx.x & 31;
    int NT = Cout >> 4, CT = Cin >> 5, NG = NT >> 2;
    long long wave = (long long)blockIdx.x * 8 + (threadIdx.x >> 5);
    int ng = (int)(wave % NG);
    long long t = wave / NG;
    int xt = (int)(t & 3);  t >>= 2;
    int y  = (int)(t & 63); t >>= 6;
    int b  = (int)t;

    int mlane = lane & 15, khalf = lane >> 4;
    v8f acc[4];
#pragma unroll
    for (int tt = 0; tt < 4; ++tt) {
        float bv = bias[ng * 64 + tt * 16 + mlane];
#pragma unroll
        for (int i = 0; i < 8; ++i) acc[tt][i] = bv;
    }

    int x = xt * 16 + mlane;                    // this lane's A-row output x
    for (int ky = 0; ky < 3; ++ky) {
        int yy = y + ky - 1;
        bool yok = (yy >= 0) && (yy < HH);
        for (int kx = 0; kx < 3; ++kx) {
            int xx = x + kx - 1;
            bool ok = yok && (xx >= 0) && (xx < WWD);
            const bf16_t* ap = in + ((long long)(b * HH + yy) * WWD + xx) * Cin;
            long long tb = (long long)((ky * 3 + kx) * CT) * NT + ng * 4;
            for (int ct = 0; ct < CT; ++ct) {
                if (ok && ct + 1 < CT) __builtin_prefetch(ap + (ct + 1) * 32, 0, 0);
                v16bf a = load_a16(ap + ct * 32, khalf, ok);
#pragma unroll
                for (int tt = 0; tt < 4; ++tt) {
                    v16bf bm = *(const v16bf*)(wp + ((tb + (long long)ct * NT + tt) * 32 + lane) * 16);
                    acc[tt] = __builtin_amdgcn_wmma_f32_16x16x32_bf16(false, a, false, bm, (short)0, acc[tt], false, false);
                }
            }
        }
    }
#pragma unroll
    for (int tt = 0; tt < 4; ++tt) {
        int n = ng * 64 + tt * 16 + mlane;
#pragma unroll
        for (int r = 0; r < 8; ++r) {
            int m = (khalf << 3) + r;
            float v = acc[tt][r];
            v = (act == 1) ? tanhf(v) : fmaxf(v, 0.f);
            int ox = xt * 16 + m;
            if (nchwF)
                outF[((long long)(b * Cout + n) * HH + y) * WWD + ox] = v;
            else
                outB[((long long)(b * HH + y) * WWD + ox) * Cout + n] = (bf16_t)v;
        }
    }
}

// ---------------- agent extraction / head ----------------
__global__ void find_cell(const int* __restrict__ mp, int* cell, long long total) {
    long long i = (long long)blockIdx.x * blockDim.x + threadIdx.x;
    if (i >= total) return;
    if (mp[i] == 0) atomicMin(&cell[(int)(i >> 12)], (int)(i & 4095));
}

__global__ void build_agent(const float* __restrict__ rm, const float* __restrict__ g,
                            const int* __restrict__ cell, float* __restrict__ out, int total) {
    int i = blockIdx.x * blockDim.x + threadIdx.x;
    if (i >= total) return;
    int f = i & 127, b = i >> 7;
    if (f < 64)
        out[i] = rm[((long long)(b * 64 + f)) * 4096 + cell[b]];
    else
        out[i] = g[((long long)b * NN) * GEc + (f - 64)];
}

__global__ void agent_head(const float* __restrict__ ab, const float* __restrict__ a1w,
                           const float* __restrict__ a1b, const float* __restrict__ a2w,
                           const float* __restrict__ a2b, float* __restrict__ res) {
    int b = threadIdx.x;
    if (b >= BB) return;
    float o = a2b[0];
    for (int i = 0; i < 32; ++i) {
        float h = a1b[i];
        for (int j = 0; j < 256; ++j) {
            float xv = (j < 128) ? ab[b * 128 + j] : ab[BB * 128 + b * 128 + (j - 128)];
            h += xv * a1w[i * 256 + j];
        }
        o += fmaxf(h, 0.f) * a2w[i];
    }
    res[b] = o;
}

// ---------------- small direct conv (DiffConv head); channel-concat aware ----------------
__global__ void conv2d_direct(const float* __restrict__ inA, const float* __restrict__ inB, int CinA,
                              const float* __restrict__ w, const float* __restrict__ bias,
                              float* __restrict__ out, int Cin, int Hi, int Wi, int Cout,
                              int Ho, int Wo, int KH, int KW, int stride, int pad, int act,
                              long long total) {
    long long i = (long long)blockIdx.x * blockDim.x + threadIdx.x;
    if (i >= total) return;
    int xo = (int)(i % Wo); long long tq = i / Wo;
    int yo = (int)(tq % Ho); tq /= Ho;
    int co = (int)(tq % Cout);
    int b  = (int)(tq / Cout);
    float acc = bias[co];
    for (int cin = 0; cin < Cin; ++cin) {
        const float* ip = (cin < CinA)
            ? inA + (long long)(b * CinA + cin) * Hi * Wi
            : inB + (long long)(b * (Cin - CinA) + (cin - CinA)) * Hi * Wi;
        const float* wp = w + ((long long)co * Cin + cin) * KH * KW;
        for (int ky = 0; ky < KH; ++ky) {
            int yi = yo * stride - pad + ky;
            if (yi < 0 || yi >= Hi) continue;
            for (int kx = 0; kx < KW; ++kx) {
                int xi = xo * stride - pad + kx;
                if (xi < 0 || xi >= Wi) continue;
                acc += ip[yi * Wi + xi] * wp[ky * KW + kx];
            }
        }
    }
    out[i] = act ? fmaxf(acc, 0.f) : acc;
}

__global__ void combine_kernel(const float* __restrict__ y4, const float* __restrict__ ares,
                               float* __restrict__ out, int total) {
    int i = blockIdx.x * blockDim.x + threadIdx.x;
    if (i >= total) return;
    int b = i / 49;
    float s1 = 1.f / (1.f + expf(-y4[i]));
    float s2 = 1.f / (1.f + expf(-ares[b]));
    out[i] = 0.5f * (s1 + s2);
}

// ---------------- host orchestration ----------------
static inline unsigned gblk(long long total, int bs) { return (unsigned)((total + bs - 1) / bs); }

extern "C" void kernel_launch(void* const* d_in, const int* in_sizes, int n_in,
                              void* d_out, int out_size, void* d_ws, size_t ws_size,
                              hipStream_t stream) {
    (void)in_sizes; (void)n_in; (void)out_size; (void)ws_size;
    const float* x[2]  = { (const float*)d_in[0], (const float*)d_in[4] };
    const int* src[2]  = { (const int*)d_in[1], (const int*)d_in[5] };
    const int* dst[2]  = { (const int*)d_in[2], (const int*)d_in[6] };
    const int* mp[2]   = { (const int*)d_in[3], (const int*)d_in[7] };
    const float *W1 = (const float*)d_in[8],  *W2 = (const float*)d_in[9],  *W3 = (const float*)d_in[10];
    const float *c1w = (const float*)d_in[11], *c1b = (const float*)d_in[12];
    const float *c2w = (const float*)d_in[13], *c2b = (const float*)d_in[14];
    const float *c3w = (const float*)d_in[15], *c3b = (const float*)d_in[16];
    const float *d1w = (const float*)d_in[17], *d1b = (const float*)d_in[18];
    const float *d2w = (const float*)d_in[19], *d2b = (const float*)d_in[20];
    const float *d3w = (const float*)d_in[21], *d3b = (const float*)d_in[22];
    const float *d4w = (const float*)d_in[23], *d4b = (const float*)d_in[24];
    const float *a1w = (const float*)d_in[25], *a1b = (const float*)d_in[26];
    const float *a2w = (const float*)d_in[27], *a2b = (const float*)d_in[28];
    float* outp = (float*)d_out;

    char* ws = (char*)d_ws;
    size_t ofs = 0;
    auto alloc = [&](size_t bytes) -> void* {
        ofs = (ofs + 255) & ~(size_t)255;
        void* p = ws + ofs;
        ofs += bytes;
        return p;
    };
    float*  degOut = (float*)alloc((size_t)NTOT * 4);
    float*  degIn  = (float*)alloc((size_t)NTOT * 4);
    bf16_t* xbf    = (bf16_t*)alloc((size_t)NTOT * DD * 2);   // bf16 GEMM input
    float*  bufY   = (float*)alloc((size_t)NTOT * DD * 4);    // GEMM out (f32)
    float*  bufZ   = (float*)alloc((size_t)NTOT * DD * 4);    // edge agg (f32)
    float*  gbuf   = (float*)alloc((size_t)NTOT * GEc * 4);   // g (f32)
    bf16_t* repmap = (bf16_t*)alloc((size_t)BB * HH * WWD * 64 * 2);
    bf16_t* c1out  = (bf16_t*)alloc((size_t)BB * HH * WWD * 256 * 2);
    bf16_t* c2out  = (bf16_t*)alloc((size_t)BB * HH * WWD * 128 * 2);
    float*  rm1    = (float*)alloc((size_t)BB * 64 * HH * WWD * 4);
    int*    cell   = (int*)  alloc((size_t)BB * 4);
    float*  agBuf  = (float*)alloc((size_t)2 * BB * 128 * 4);
    float*  agRes  = (float*)alloc((size_t)BB * 4);
    float*  y1     = (float*)alloc((size_t)32 * 32 * 62 * 62 * 4);
    float*  y2     = (float*)alloc((size_t)32 * 16 * 31 * 31 * 4);
    float*  y3     = (float*)alloc((size_t)32 * 8  * 15 * 15 * 4);
    float*  y4     = (float*)alloc((size_t)32 * 49 * 4);
    bf16_t* W1p = (bf16_t*)alloc((size_t)4 * 8 * 512 * 2);
    bf16_t* W2p = (bf16_t*)alloc((size_t)4 * 8 * 512 * 2);
    bf16_t* W3p = (bf16_t*)alloc((size_t)4 * 4 * 512 * 2);
    bf16_t* c1p = (bf16_t*)alloc((size_t)9 * 2 * 16 * 512 * 2);
    bf16_t* c2p = (bf16_t*)alloc((size_t)9 * 8 * 8  * 512 * 2);
    bf16_t* c3p = (bf16_t*)alloc((size_t)9 * 4 * 4  * 512 * 2);

    float* rm2 = outp + 1568;   // second output region (rm2, NCHW [32,64,64,64])

    // ---- pack weights into WMMA bf16 layouts ----
    pack_b_bf16<<<4 * 8, 32, 0, stream>>>(W1, W1p, 128, 128);
    pack_b_bf16<<<4 * 8, 32, 0, stream>>>(W2, W2p, 128, 128);
    pack_b_bf16<<<4 * 4, 32, 0, stream>>>(W3, W3p, 128, 64);
    pack_conv_bf16<<<9 * 2 * 16, 32, 0, stream>>>(c1w, c1p, 64, 256);
    pack_conv_bf16<<<9 * 8 * 8,  32, 0, stream>>>(c2w, c2p, 256, 128);
    pack_conv_bf16<<<9 * 4 * 4,  32, 0, stream>>>(c3w, c3p, 128, 64);

    const long long nf128 = (long long)NTOT * 128;
    const long long nf64  = (long long)NTOT * 64;

    for (int s = 0; s < 2; ++s) {
        float* rmOut = (s == 0) ? rm1 : rm2;
        // degrees
        fillf_kernel<<<gblk(NTOT, 256), 256, 0, stream>>>(degOut, 0.f, NTOT);
        fillf_kernel<<<gblk(NTOT, 256), 256, 0, stream>>>(degIn, 0.f, NTOT);
        edge_deg<<<gblk(NE, 256), 256, 0, stream>>>(src[s], dst[s], degOut, degIn, NE);

        // gconv layer 1 (relu)
        f32_to_bf16_kernel<<<gblk(nf128, 256), 256, 0, stream>>>(x[s], xbf, nf128);
        gemm_wmma<<<(NTOT / 16) * 2 / 8, 256, 0, stream>>>(xbf, W1p, bufY, NTOT, 128, 128);
        fillf_kernel<<<gblk(nf128, 256), 256, 0, stream>>>(bufZ, 0.f, nf128);
        scatter_edges<<<gblk((long long)NE * 128, 256), 256, 0, stream>>>(bufY, src[s], dst[s], degOut, bufZ, 128, (long long)NE * 128);
        scale_relu_bf16<<<gblk(nf128, 256), 256, 0, stream>>>(bufZ, degIn, xbf, 128, nf128);
        // gconv layer 2 (relu)
        gemm_wmma<<<(NTOT / 16) * 2 / 8, 256, 0, stream>>>(xbf, W2p, bufY, NTOT, 128, 128);
        fillf_kernel<<<gblk(nf128, 256), 256, 0, stream>>>(bufZ, 0.f, nf128);
        scatter_edges<<<gblk((long long)NE * 128, 256), 256, 0, stream>>>(bufY, src[s], dst[s], degOut, bufZ, 128, (long long)NE * 128);
        scale_relu_bf16<<<gblk(nf128, 256), 256, 0, stream>>>(bufZ, degIn, xbf, 128, nf128);
        // gconv layer 3 (tanh) -> g (f32)
        gemm_wmma<<<(NTOT / 16) * 1 / 8, 256, 0, stream>>>(xbf, W3p, bufY, NTOT, 128, 64);
        fillf_kernel<<<gblk(nf64, 256), 256, 0, stream>>>(bufZ, 0.f, nf64);
        scatter_edges<<<gblk((long long)NE * 64, 256), 256, 0, stream>>>(bufY, src[s], dst[s], degOut, bufZ, 64, (long long)NE * 64);
        scale_tanh_f32<<<gblk(nf64, 256), 256, 0, stream>>>(bufZ, degIn, gbuf, 64, nf64);

        // rep map (NHWC bf16) and the 3 WMMA convs
        build_repmap<<<gblk((long long)BB * 4096 * 64, 256), 256, 0, stream>>>(gbuf, mp[s], repmap, (long long)BB * 4096 * 64);
        conv3x3_wmma<<<(BB * HH * 4 * 4) / 8, 256, 0, stream>>>(repmap, c1p, c1b, c1out, nullptr, 64, 256, 0, 0);
        conv3x3_wmma<<<(BB * HH * 4 * 2) / 8, 256, 0, stream>>>(c1out, c2p, c2b, c2out, nullptr, 256, 128, 0, 0);
        conv3x3_wmma<<<(BB * HH * 4 * 1) / 8, 256, 0, stream>>>(c2out, c3p, c3b, nullptr, rmOut, 128, 64, 1, 1);

        // agent features
        filli_kernel<<<1, BB, 0, stream>>>(cell, 1 << 30, BB);
        find_cell<<<gblk((long long)BB * 4096, 256), 256, 0, stream>>>(mp[s], cell, (long long)BB * 4096);
        build_agent<<<gblk(BB * 128, 128), 128, 0, stream>>>(rmOut, gbuf, cell, agBuf + s * BB * 128, BB * 128);
    }

    // agent head
    agent_head<<<1, BB, 0, stream>>>(agBuf, a1w, a1b, a2w, a2b, agRes);

    // DiffConv head on concat([rm1, rm2], ch)
    long long t1 = (long long)32 * 32 * 62 * 62;
    conv2d_direct<<<gblk(t1, 256), 256, 0, stream>>>(rm1, rm2, 64, d1w, d1b, y1, 128, 64, 64, 32, 62, 62, 3, 3, 1, 0, 1, t1);
    long long t2 = (long long)32 * 16 * 31 * 31;
    conv2d_direct<<<gblk(t2, 256), 256, 0, stream>>>(y1, y1, 32, d2w, d2b, y2, 32, 62, 62, 16, 31, 31, 4, 4, 2, 1, 1, t2);
    long long t3 = (long long)32 * 8 * 15 * 15;
    conv2d_direct<<<gblk(t3, 256), 256, 0, stream>>>(y2, y2, 16, d3w, d3b, y3, 16, 31, 31, 8, 15, 15, 4, 4, 2, 1, 1, t3);
    long long t4 = (long long)32 * 1 * 7 * 7;
    conv2d_direct<<<gblk(t4, 256), 256, 0, stream>>>(y3, y3, 8, d4w, d4b, y4, 8, 15, 15, 1, 7, 7, 4, 4, 2, 1, 0, t4);

    combine_kernel<<<gblk(1568, 256), 256, 0, stream>>>(y4, agRes, outp, 1568);
}
